// AttentionNet_68101001445571
// MI455X (gfx1250) — matrix operand order repaired and checked by reading, not compiled, not run on hardware
//
#include <hip/hip_runtime.h>
#include <hip/hip_bf16.h>
#include <stdint.h>

// Problem constants (match reference)
#define BB 256
#define TT 128
#define NN 512
#define HH 64
#define DU 6
#define DS 6
#define CC 10.0f
#define NEGV (-103.27892990343184f)   // log(1e-45)

typedef __attribute__((ext_vector_type(16))) _Float16 v16h;
typedef __attribute__((ext_vector_type(8)))  float    v8f;

__device__ __forceinline__ v8f wmma_f16(v16h a, v16h b, v8f c) {
  // (neg_a, A, neg_b, B, c_mod, C, reuse_a, reuse_b)
  return __builtin_amdgcn_wmma_f32_16x16x32_f16(false, a, false, b, (short)0, c,
                                                false, false);
}

__device__ __forceinline__ float fast_tanh(float x) {
  float y;
  asm volatile("v_tanh_f32 %0, %1" : "=v"(y) : "v"(x));
  return y;
}

// Shared WMMA stage: out[rowbase + 0..127][0..63] = Ash(128x64,f16) @ Bsh(64x64,f16)
// Block = 256 threads = 8 waves, one 16-row M-tile per wave, K=64 as 2x K32 chunks.
__device__ __forceinline__ void wmma_128x64x64(const _Float16* __restrict__ Ash,
                                               const _Float16* __restrict__ Bsh,
                                               float* __restrict__ outg,
                                               size_t rowbase) {
  const int wave = threadIdx.x >> 5;
  const int lane = threadIdx.x & 31;
  const int m    = lane & 15;
  const int hi   = lane >> 4;

  // A fragments: lane holds row m of the wave's tile; 16 halfs with ISA K mapping.
  v16h a0, a1;
  const int arow = wave * 16 + m;
#pragma unroll
  for (int e = 0; e < 16; ++e) {
    const int k = (e < 8 ? e : e + 8) + hi * 8;     // K within a 32-chunk
    a0[e] = Ash[arow * 64 + k];
    a1[e] = Ash[arow * 64 + 32 + k];
  }

  // B: lane indexes N (column m), elems index K with ISA K mapping
#pragma unroll
  for (int nt = 0; nt < 4; ++nt) {
    v16h b0, b1;
#pragma unroll
    for (int e = 0; e < 16; ++e) {
      const int kb = (e < 8 ? e : e + 8) + hi * 8;
      b0[e] = Bsh[kb * 64 + nt * 16 + m];
      b1[e] = Bsh[(kb + 32) * 64 + nt * 16 + m];
    }
    v8f c = {};
    c = wmma_f16(a0, b0, c);
    c = wmma_f16(a1, b1, c);
#pragma unroll
    for (int v = 0; v < 8; ++v) {
      const int row = wave * 16 + v + hi * 8;        // C/D: VGPR v -> M = v + 8*hi
      const int col = nt * 16 + m;                   // N striped over lanes 0..15
      outg[(rowbase + row) * 64 + col] = c[v];
    }
  }
}

// -------- Kernel A: dec_all[b,t,:] = (user_inp @ W_user + b_user) @ W2 --------
__global__ void __launch_bounds__(256)
k_dec_all(const float* __restrict__ user_seq, const float* __restrict__ W_user,
          const float* __restrict__ b_user, const float* __restrict__ W2,
          float* __restrict__ dec_all) {
  __shared__ _Float16 ueh[128 * 64];   // uemb tile, f16
  __shared__ _Float16 w2h[64 * 64];    // W2, f16
  __shared__ float    wu[DU * 64];
  __shared__ float    bu[64];

  const int tid = threadIdx.x;
  for (int e = tid; e < 64 * 64; e += 256) w2h[e] = (_Float16)W2[e];
  for (int e = tid; e < DU * 64; e += 256) wu[e] = W_user[e];
  if (tid < 64) bu[tid] = b_user[tid];
  __syncthreads();

  const size_t rowbase = (size_t)blockIdx.x * 128;   // row index into B*T
  for (int e = tid; e < 128 * 64; e += 256) {
    const int row = e >> 6, h = e & 63;
    const size_t r = rowbase + row;
    float acc = bu[h];
#pragma unroll
    for (int i = 0; i < DU; ++i) acc += user_seq[r * 7 + i] * wu[i * 64 + h];
    ueh[e] = (_Float16)acc;
  }
  __syncthreads();
  wmma_128x64x64(ueh, w2h, dec_all, rowbase);
}

// -------- Kernel B: enc0[b,n,:] = ((static,cap0,0) @ W_srv + b_srv) @ W1; init cap --------
__global__ void __launch_bounds__(256)
k_init_enc(const float* __restrict__ server_seq, const float* __restrict__ W_srv,
           const float* __restrict__ b_srv, const float* __restrict__ W1,
           float* __restrict__ enc, float* __restrict__ capw) {
  __shared__ _Float16 hh[128 * 64];
  __shared__ _Float16 w1h[64 * 64];
  __shared__ float    ws[7 * 64];
  __shared__ float    bs[64];

  const int tid = threadIdx.x;
  for (int e = tid; e < 64 * 64; e += 256) w1h[e] = (_Float16)W1[e];
  for (int e = tid; e < 7 * 64; e += 256) ws[e] = W_srv[e];
  if (tid < 64) bs[tid] = b_srv[tid];
  __syncthreads();

  const size_t rowbase = (size_t)blockIdx.x * 128;   // row index into B*N
  for (int e = tid; e < 128 * 64; e += 256) {
    const int row = e >> 6, j = e & 63;
    const size_t g = rowbase + row;
    float acc = bs[j];                                // active = 0 -> last term drops
#pragma unroll
    for (int i = 0; i < DS; ++i) acc += server_seq[g * 6 + i] * ws[i * 64 + j];
    hh[e] = (_Float16)acc;
  }
  // cap0 init: 128 rows x 4
  for (int e = tid; e < 128 * 4; e += 256) {
    const int row = e >> 2, c = e & 3;
    const size_t g = rowbase + row;
    capw[g * 4 + c] = server_seq[g * 6 + 2 + c];
  }
  __syncthreads();
  wmma_128x64x64(hh, w1h, enc, rowbase);
}

// -------- Kernel C: sequential pointer scan, one workgroup per batch --------
__global__ void __launch_bounds__(256)
k_scan(const float* __restrict__ user_seq, const float* __restrict__ server_seq,
       const uint8_t* __restrict__ masks, const float* __restrict__ W_srv,
       const float* __restrict__ b_srv, const float* __restrict__ W1,
       const float* __restrict__ vt, const float* __restrict__ dec_all,
       float* __restrict__ enc, float* __restrict__ capw,
       float* __restrict__ out) {
  __shared__ float w1_sh[64 * 64];
  __shared__ float wsrv_sh[7 * 64];
  __shared__ float bsrv_sh[64];
  __shared__ float vt_sh[64];
  __shared__ float dec_sh[64];
  __shared__ float u_sh[NN];
  __shared__ float red_val[256];
  __shared__ int   red_idx[256];
  __shared__ float red_sum[256];
  __shared__ float h_sh[64];
  __shared__ float sseq_sh[8];

  const int tid = threadIdx.x;
  const int b   = blockIdx.x;

  for (int e = tid; e < 64 * 64; e += 256) w1_sh[e] = W1[e];
  for (int e = tid; e < 7 * 64; e += 256) wsrv_sh[e] = W_srv[e];
  if (tid < 64) { bsrv_sh[tid] = b_srv[tid]; vt_sh[tid] = vt[tid]; }
  __syncthreads();

  const float* encb = enc + (size_t)b * NN * 64;
  const int n0 = tid, n1 = tid + 256;

  for (int t = 0; t < TT; ++t) {
    // stage dec_t; prefetch next step's row into cache
    if (tid < 64) dec_sh[tid] = dec_all[((size_t)b * TT + t) * 64 + tid];
    if (tid == 0 && t + 1 < TT)
      __builtin_prefetch(&dec_all[((size_t)b * TT + t + 1) * 64], 0, 1);
    __syncthreads();

    // u[n] = tanh(enc[n]+dec) . vt   (2 rows per thread)
    const uint8_t* mrow = masks + ((size_t)b * TT + t) * NN;
    float u0, u1;
    {
      const float* er = encb + (size_t)n0 * 64;
      float acc = 0.f;
#pragma unroll 8
      for (int h = 0; h < 64; ++h) acc += fast_tanh(er[h] + dec_sh[h]) * vt_sh[h];
      u0 = mrow[n0] ? acc : NEGV;
      u_sh[n0] = u0;
    }
    {
      const float* er = encb + (size_t)n1 * 64;
      float acc = 0.f;
#pragma unroll 8
      for (int h = 0; h < 64; ++h) acc += fast_tanh(er[h] + dec_sh[h]) * vt_sh[h];
      u1 = mrow[n1] ? acc : NEGV;
      u_sh[n1] = u1;
    }

    // argmax/max reduction (ties -> lowest index, matching jnp.argmax)
    float bv; int bi;
    if (u0 >= u1) { bv = u0; bi = n0; } else { bv = u1; bi = n1; }
    red_val[tid] = bv; red_idx[tid] = bi;
    __syncthreads();
#pragma unroll
    for (int s = 128; s > 0; s >>= 1) {
      if (tid < s) {
        const float ov = red_val[tid + s]; const int oi = red_idx[tid + s];
        if (ov > red_val[tid] || (ov == red_val[tid] && oi < red_idx[tid])) {
          red_val[tid] = ov; red_idx[tid] = oi;
        }
      }
      __syncthreads();
    }
    const float umax = red_val[0];
    const int   sel  = red_idx[0];

    // p = 1 / sum_n exp(C*(u_n - umax))
    red_sum[tid] = __expf(CC * (u_sh[n0] - umax)) + __expf(CC * (u_sh[n1] - umax));
    __syncthreads();
#pragma unroll
    for (int s = 128; s > 0; s >>= 1) {
      if (tid < s) red_sum[tid] += red_sum[tid + s];
      __syncthreads();
    }
    if (tid == 0) {
      out[(size_t)t * BB + b]                      = 1.0f / red_sum[0];   // probs (T,B)
      out[(size_t)TT * BB + (size_t)b * TT + t]    = (float)sel;          // idx   (B,T)
    }

    // incremental state update for the selected server row
    if (tid < 8) {
      if (tid < 2) {
        sseq_sh[tid] = server_seq[((size_t)b * NN + sel) * 6 + tid];
      } else if (tid < 6) {
        const int c = tid - 2;
        const float dem = user_seq[((size_t)b * TT + t) * 7 + 2 + c];
        const size_t ci = ((size_t)b * NN + sel) * 4 + c;
        const float nc = capw[ci] - dem;
        capw[ci] = nc;
        sseq_sh[tid] = nc;
      } else if (tid == 6) {
        sseq_sh[6] = 1.0f;                          // alloc>0 after this step
      }
    }
    __syncthreads();
    if (tid < 64) {
      float acc = bsrv_sh[tid];
#pragma unroll
      for (int i = 0; i < 7; ++i) acc += sseq_sh[i] * wsrv_sh[i * 64 + tid];
      h_sh[tid] = acc;
    }
    __syncthreads();
    if (tid < 64) {
      float acc = 0.f;
#pragma unroll 8
      for (int j = 0; j < 64; ++j) acc += h_sh[j] * w1_sh[j * 64 + tid];
      enc[((size_t)b * NN + sel) * 64 + tid] = acc;
    }
    __threadfence();
    __syncthreads();
  }
}

extern "C" void kernel_launch(void* const* d_in, const int* in_sizes, int n_in,
                              void* d_out, int out_size, void* d_ws, size_t ws_size,
                              hipStream_t stream) {
  const float*   user_seq   = (const float*)d_in[0];
  const float*   server_seq = (const float*)d_in[1];
  const uint8_t* masks      = (const uint8_t*)d_in[2];
  const float*   W_user     = (const float*)d_in[3];
  const float*   b_user     = (const float*)d_in[4];
  const float*   W_srv      = (const float*)d_in[5];
  const float*   b_srv      = (const float*)d_in[6];
  const float*   W1         = (const float*)d_in[7];
  const float*   W2         = (const float*)d_in[8];
  const float*   vt         = (const float*)d_in[9];
  float* out = (float*)d_out;

  // workspace layout
  float* dec_all = (float*)d_ws;                           // B*T*H
  float* enc     = dec_all + (size_t)BB * TT * HH;         // B*N*H
  float* capw    = enc + (size_t)BB * NN * HH;             // B*N*4

  // A: dec_all (32768 rows, 128 rows/block)
  k_dec_all<<<dim3((BB * TT) / 128), dim3(256), 0, stream>>>(user_seq, W_user,
                                                             b_user, W2, dec_all);
  // B: enc0 + cap init (131072 rows, 128 rows/block)
  k_init_enc<<<dim3((BB * NN) / 128), dim3(256), 0, stream>>>(server_seq, W_srv,
                                                              b_srv, W1, enc, capw);
  // C: sequential scan, one workgroup per batch
  k_scan<<<dim3(BB), dim3(256), 0, stream>>>(user_seq, server_seq, masks, W_srv,
                                             b_srv, W1, vt, dec_all, enc, capw, out);
}
